// DCMBlock_26903675142415
// MI455X (gfx1250) — matrix operand order repaired and checked by reading, not compiled, not run on hardware
//
#include <hip/hip_runtime.h>
#include <math.h>

// ---------------------------------------------------------------------------
// Problem constants (B, CIN, CMID, H, W, K, PAD) = (2, 192, 180, 96, 96, 3, 1)
// ---------------------------------------------------------------------------
#define BDIM   2
#define CIN_C  192
#define CMID_C 180
#define HDIM   96
#define WDIM   96
#define HW     (HDIM * WDIM)        // 9216
#define KK     9                    // 3x3 taps
#define CG     18                   // CMID / groups(10)
#define KDEF   (CMID_C * KK)        // 1620
#define KDEFP  1632                 // 1620 padded to mult of 32
#define CMIDP  192                  // 180 padded to mult of 16/32
#define COFFP  368                  // 360 padded to mult of 16

typedef __attribute__((ext_vector_type(16))) __bf16 v16bf;
typedef __attribute__((ext_vector_type(8)))  __bf16 v8bf;
typedef __attribute__((ext_vector_type(8)))  float  v8f;

__device__ __forceinline__ __bf16 to_bf16(float f) { return (__bf16)f; }

// ---------------------------------------------------------------------------
// Zero-padded weight/bias packer:  Wp[MP,KP] bf16 = pad(W[M,K]); bdst[MP] f32
// Zero pad rows/cols => padded output rows of every GEMM are exactly 0, which
// zero-pads the K dimension of the consumer GEMM for free.
// ---------------------------------------------------------------------------
__global__ __launch_bounds__(256) void pack_weight(
    const float* __restrict__ W, const float* __restrict__ bsrc,
    __bf16* __restrict__ Wp, float* __restrict__ bdst,
    int M, int K, int MP, int KP)
{
    long tid = (long)blockIdx.x * blockDim.x + threadIdx.x;
    if (tid >= (long)MP * KP) return;
    const int m = (int)(tid / KP);
    const int k = (int)(tid % KP);
    const float v = (m < M && k < K) ? W[(long)m * K + k] : 0.f;
    Wp[tid] = (__bf16)v;
    if (k == 0) bdst[m] = (m < M) ? bsrc[m] : 0.f;
}

// ---------------------------------------------------------------------------
// Wave-level WMMA GEMM for the 1x1 convs (f32 B sources), branch-free.
//   C[b, m, n] = sum_k A[m,k] * B_b[k,n] + bias[m]
// MODE 0: B = f32 matrix [KP, HW] per batch (cvt to bf16 on load)
// MODE 2: B = bf16(attn0*Fh + attn1*Fw) fused on the fly (expand conv)
// Each wave owns a 16x64 tile: 4 accumulators, A fragment reused 4x per step.
// ---------------------------------------------------------------------------
template <int MP, int KP, int MODE>
__global__ __launch_bounds__(256) void wmma_gemm(
    const __bf16* __restrict__ A,
    const float*  __restrict__ Bf,
    const float*  __restrict__ Fh,
    const float*  __restrict__ Fw,
    const float*  __restrict__ attn,   // [B*2]
    const float*  __restrict__ bias,   // [MP] (zero-padded)
    float*        __restrict__ C,      // [B, MP, HW]
    long bsB)
{
    constexpr int numMT = MP / 16;
    constexpr int numNT = HW / 64;                 // 144
    constexpr int total = BDIM * numMT * numNT;

    const int lane = threadIdx.x & 31;
    const int wave = blockIdx.x * (blockDim.x >> 5) + (threadIdx.x >> 5);
    if (wave >= total) return;

    const int b  = wave / (numMT * numNT);
    const int r  = wave % (numMT * numNT);
    const int m0 = (r / numNT) * 16;
    const int n0 = (r % numNT) * 64;

    float a0 = 0.f, a1 = 0.f;
    if (MODE == 2) { a0 = attn[b * 2 + 0]; a1 = attn[b * 2 + 1]; }

    const float* Bf_b = (MODE == 0) ? Bf + (long)b * bsB : nullptr;
    const float* Fh_b = (MODE == 2) ? Fh + (long)b * bsB : nullptr;
    const float* Fw_b = (MODE == 2) ? Fw + (long)b * bsB : nullptr;

    const v8f vzero = {0.f, 0.f, 0.f, 0.f, 0.f, 0.f, 0.f, 0.f};
    v8f acc[4] = {vzero, vzero, vzero, vzero};

    const int half  = lane >> 4;
    const int mlane = lane & 15;
    const int mrow  = m0 + mlane;                  // always < MP (padded)

    #pragma unroll 2
    for (int k0 = 0; k0 < KP; k0 += 32) {
        const __bf16* arow = A + (long)mrow * KP + k0 + (half << 3);
        union { v16bf v; v8bf h[2]; } au;
        au.h[0] = *(const v8bf*)(arow);            // k = k0 + half*8 + 0..7
        au.h[1] = *(const v8bf*)(arow + 16);       // k = k0+16 + half*8 + 0..7
        const v16bf afrag = au.v;

        const long krow = (long)k0 * HW;           // B k-step base row

        #pragma unroll
        for (int s = 0; s < 4; ++s) {
            const int nbase = n0 + s * 16;
            v16bf bfrag;
            if (MODE == 0) {
                const float* src = Bf_b + krow + nbase + mlane;
                #pragma unroll
                for (int e = 0; e < 8; ++e) {
                    bfrag[e]     = to_bf16(src[(long)((half << 3) + e) * HW]);
                    bfrag[e + 8] = to_bf16(src[(long)(16 + (half << 3) + e) * HW]);
                }
            } else {
                const float* sh = Fh_b + krow + nbase + mlane;
                const float* sw = Fw_b + krow + nbase + mlane;
                #pragma unroll
                for (int e = 0; e < 8; ++e) {
                    const long o0 = (long)((half << 3) + e) * HW;
                    const long o1 = (long)(16 + (half << 3) + e) * HW;
                    bfrag[e]     = to_bf16(a0 * sh[o0] + a1 * sw[o0]);
                    bfrag[e + 8] = to_bf16(a0 * sh[o1] + a1 * sw[o1]);
                }
            }
            acc[s] = __builtin_amdgcn_wmma_f32_16x16x32_bf16(
                false, afrag, false, bfrag, (short)0, acc[s], false, false);
        }
    }

    float bv[8];
    #pragma unroll
    for (int rr = 0; rr < 8; ++rr) bv[rr] = bias[m0 + (half << 3) + rr];
    #pragma unroll
    for (int s = 0; s < 4; ++s) {
        const int n = n0 + s * 16 + mlane;
        #pragma unroll
        for (int rr = 0; rr < 8; ++rr) {
            const int m = m0 + (half << 3) + rr;
            C[((long)b * MP + m) * HW + n] = acc[s][rr] + bv[rr];
        }
    }
}

// ---------------------------------------------------------------------------
// Deform GEMM (the dominant cost, K = 1632): block-cooperative with async
// global->LDS double-buffered B staging (CDNA5 ASYNCcnt path).
//   Block (8 waves) tile: 64M x 128N; waves = 4 m-tiles x 2 n-chunks.
//   B slab: 32 k-rows x 128 n bf16 in LDS (row stride 272B to spread banks),
//   filled by 512 global_load_async_to_lds_b128 chunks (2 per thread),
//   double buffered, synced with s_wait_asynccnt + workgroup barriers.
// The asm's LDS-address operand is derived from the *real* generic pointer
// into smemB (low 32 bits of a generic LDS pointer == LDS byte address), so
// the array escapes into the asm and the compiler must keep the ds_load reads.
// ---------------------------------------------------------------------------
#define BROW 136                       // 128 + 8 bf16 pad (272 B row stride)
__global__ __launch_bounds__(256) void wmma_gemm_deform(
    const __bf16* __restrict__ A,      // [CMIDP, KDEFP] packed bf16
    const __bf16* __restrict__ Bb,     // [B, KDEFP, HW] bf16 patches
    const float*  __restrict__ bias,   // [CMIDP]
    float*        __restrict__ C)      // [B, CMIDP, HW]
{
    __shared__ __bf16 smemB[2][32][BROW];          // 2 x 8704 B

    constexpr int mblkCnt = CMIDP / 64;            // 3
    constexpr int nblkCnt = HW / 128;              // 72
    const int blk  = blockIdx.x;
    const int b    = blk / (mblkCnt * nblkCnt);
    const int rem  = blk % (mblkCnt * nblkCnt);
    const int mblk = rem / nblkCnt;
    const int nblk = rem % nblkCnt;
    const int nb0  = nblk * 128;

    const int t    = threadIdx.x;
    const int lane = t & 31;
    const int w    = t >> 5;
    const int m0   = mblk * 64 + (w & 3) * 16;     // wave's m-tile
    const int nloc = (w >> 2) * 64;                // wave's n-chunk in block

    const __bf16* Bb_b = Bb + (long)b * KDEFP * HW;

    // async fill of one 32x128 bf16 slab for k-rows [kbase, kbase+32)
    auto fill = [&](int buf, int kbase) {
        #pragma unroll
        for (int c2 = 0; c2 < 2; ++c2) {
            const int c   = t + c2 * 256;          // chunk 0..511 (16B each)
            const int row = c >> 4;                // k-row 0..31
            const int col = c & 15;                // 16B column chunk
            const __bf16* g = Bb_b + (long)(kbase + row) * HW + nb0 + col * 8;
            // real LDS byte address of the destination chunk (captures smemB)
            const unsigned lds = (unsigned)(size_t)&smemB[buf][row][col * 8];
            asm volatile("global_load_async_to_lds_b128 %0, %1, off"
                         :: "v"(lds), "v"(g) : "memory");
        }
    };

    const v8f vzero = {0.f, 0.f, 0.f, 0.f, 0.f, 0.f, 0.f, 0.f};
    v8f acc[4] = {vzero, vzero, vzero, vzero};

    const int half  = lane >> 4;
    const int mlane = lane & 15;
    const int mrow  = m0 + mlane;

    fill(0, 0);
    constexpr int KSTEPS = KDEFP / 32;             // 51
    for (int ks = 0; ks < KSTEPS; ++ks) {
        const int k0  = ks * 32;
        const int buf = ks & 1;
        if (ks + 1 < KSTEPS) {
            fill(buf ^ 1, k0 + 32);
            // 2 async ops just issued for next buf; wait for current buf's 2
            asm volatile("s_wait_asynccnt 0x2" ::: "memory");
        } else {
            asm volatile("s_wait_asynccnt 0x0" ::: "memory");
        }
        __syncthreads();                           // all waves' fills visible

        // A fragment: two aligned 16B bf16 loads straight from global
        const __bf16* arow = A + (long)mrow * KDEFP + k0 + (half << 3);
        union { v16bf v; v8bf h[2]; } au;
        au.h[0] = *(const v8bf*)(arow);
        au.h[1] = *(const v8bf*)(arow + 16);
        const v16bf afrag = au.v;

        #pragma unroll
        for (int s = 0; s < 4; ++s) {
            const __bf16* bp = &smemB[buf][0][nloc + s * 16 + mlane];
            union { v16bf v; v8bf h[2]; } bu;
            #pragma unroll
            for (int e = 0; e < 8; ++e) {
                bu.v[e]     = bp[((half << 3) + e) * BROW];
                bu.v[e + 8] = bp[(16 + (half << 3) + e) * BROW];
            }
            acc[s] = __builtin_amdgcn_wmma_f32_16x16x32_bf16(
                false, afrag, false, bu.v, (short)0, acc[s], false, false);
        }
        __syncthreads();                           // reads done before refill
    }

    float bv[8];
    #pragma unroll
    for (int rr = 0; rr < 8; ++rr) bv[rr] = bias[m0 + (half << 3) + rr];
    #pragma unroll
    for (int s = 0; s < 4; ++s) {
        const int n = nb0 + nloc + s * 16 + mlane;
        #pragma unroll
        for (int rr = 0; rr < 8; ++rr) {
            const int m = m0 + (half << 3) + rr;
            C[((long)b * CMIDP + m) * HW + n] = acc[s][rr] + bv[rr];
        }
    }
}

// ---------------------------------------------------------------------------
// Deformable im2col: patches[b, i*9+kk, p] = bilinear(xr[b,i], p+tap+off)
// offsets in offs[b, chan_base + g*18 + kk*2 + {0,1}, p] (row stride COFFP)
// Rows >= KDEF are zero-filled padding.
// ---------------------------------------------------------------------------
__global__ __launch_bounds__(256) void build_patches(
    const float* __restrict__ xr,     // [B, CMIDP, HW]
    const float* __restrict__ offs,   // [B, COFFP, HW]
    int chan_base,                    // 0 -> off_w, CMID -> off_h
    __bf16*      __restrict__ patches)// [B, KDEFP, HW]
{
    const long per_b = (long)KDEFP * HW;
    long tid = (long)blockIdx.x * blockDim.x + threadIdx.x;
    if (tid >= (long)BDIM * per_b) return;

    const int  b   = (int)(tid / per_b);
    const long rem = tid % per_b;
    const int  row = (int)(rem / HW);     // i*9 + kk  (or pad row)
    const int  p   = (int)(rem % HW);

    if (row >= KDEF) { patches[tid] = to_bf16(0.f); return; }

    const int i  = row / KK;
    const int kk = row % KK;
    const int g  = i / CG;
    const int y  = p / WDIM;
    const int x  = p % WDIM;

    const float* ob = offs + ((long)b * COFFP + chan_base) * HW;
    const float oy = ob[(long)(g * 18 + kk * 2 + 0) * HW + p];
    const float ox = ob[(long)(g * 18 + kk * 2 + 1) * HW + p];

    const float sy = oy + (float)(kk / 3 - 1) + (float)y;
    const float sx = ox + (float)(kk % 3 - 1) + (float)x;
    const float y0f = floorf(sy);
    const float x0f = floorf(sx);
    const float dy = sy - y0f;
    const float dx = sx - x0f;
    const int y0 = (int)y0f;
    const int x0 = (int)x0f;

    const float* xc = xr + ((long)b * CMIDP + i) * HW;
    float acc = 0.f;
    auto samp = [&](int yi, int xi, float wgt) {
        if (yi >= 0 && yi < HDIM && xi >= 0 && xi < WDIM)
            acc += wgt * xc[yi * WDIM + xi];
    };
    samp(y0,     x0,     (1.f - dy) * (1.f - dx));
    samp(y0,     x0 + 1, (1.f - dy) * dx);
    samp(y0 + 1, x0,     dy * (1.f - dx));
    samp(y0 + 1, x0 + 1, dy * dx);

    patches[tid] = to_bf16(acc);
}

// ---------------------------------------------------------------------------
// favg[b*CMID+cm] = mean over HW of (f_w + f_h); fw/fh row stride CMIDP.
// ---------------------------------------------------------------------------
__global__ __launch_bounds__(256) void mean_fwfh(
    const float* __restrict__ fw, const float* __restrict__ fh,
    float* __restrict__ favg)
{
    const int bc = blockIdx.x;                 // b*CMID + cm
    const int b  = bc / CMID_C;
    const int cm = bc % CMID_C;
    const float* a = fw + ((long)b * CMIDP + cm) * HW;
    const float* c = fh + ((long)b * CMIDP + cm) * HW;
    float s = 0.f;
    for (int p = threadIdx.x; p < HW; p += blockDim.x) s += a[p] + c[p];
    __shared__ float sm[256];
    sm[threadIdx.x] = s;
    __syncthreads();
    for (int off = 128; off > 0; off >>= 1) {
        if ((int)threadIdx.x < off) sm[threadIdx.x] += sm[threadIdx.x + off];
        __syncthreads();
    }
    if (threadIdx.x == 0) favg[bc] = sm[0] / (float)HW;
}

// ---------------------------------------------------------------------------
// attn[b*2+j] = softmax_j( w_attn[j,:] . favg[b,:] + b_attn[j] )
// ---------------------------------------------------------------------------
__global__ __launch_bounds__(64) void attn_kernel(
    const float* __restrict__ favg, const float* __restrict__ w_attn,
    const float* __restrict__ b_attn, float* __restrict__ attn)
{
    __shared__ float logits[BDIM][2];
    const int t = threadIdx.x;
    if (t < BDIM * 2) {
        const int b = t >> 1, j = t & 1;
        float s = b_attn[j];
        for (int c = 0; c < CMID_C; ++c)
            s += w_attn[j * CMID_C + c] * favg[b * CMID_C + c];
        logits[b][j] = s;
    }
    __syncthreads();
    if (t < BDIM) {
        const float l0 = logits[t][0], l1 = logits[t][1];
        const float m  = fmaxf(l0, l1);
        const float e0 = __expf(l0 - m), e1 = __expf(l1 - m);
        const float inv = 1.f / (e0 + e1);
        attn[t * 2 + 0] = e0 * inv;
        attn[t * 2 + 1] = e1 * inv;
    }
}

// ---------------------------------------------------------------------------
// Host-side orchestration
// ---------------------------------------------------------------------------
static inline int ceil_div(long a, long b) { return (int)((a + b - 1) / b); }

extern "C" void kernel_launch(void* const* d_in, const int* in_sizes, int n_in,
                              void* d_out, int out_size, void* d_ws, size_t ws_size,
                              hipStream_t stream) {
    const float* x        = (const float*)d_in[0];
    const float* w_reduce = (const float*)d_in[1];   // [180,192]
    const float* b_reduce = (const float*)d_in[2];
    const float* w_offset = (const float*)d_in[3];   // [360,180]
    const float* b_offset = (const float*)d_in[4];
    const float* w_dcnw   = (const float*)d_in[5];   // [180,180,3,3] -> [180,1620]
    const float* b_dcnw   = (const float*)d_in[6];
    const float* w_dcnh   = (const float*)d_in[7];
    const float* b_dcnh   = (const float*)d_in[8];
    const float* w_expand = (const float*)d_in[9];   // [192,180]
    const float* b_expand = (const float*)d_in[10];
    const float* w_attn   = (const float*)d_in[11];  // [2,180]
    const float* b_attn   = (const float*)d_in[12];
    float* out = (float*)d_out;                      // [2,192,96,96]

    // ---- scratch layout, every region 256B aligned ----
    char* wp = (char*)d_ws;
    auto alloc = [&](size_t bytes) -> char* {
        char* r = wp; wp += (bytes + 255) & ~(size_t)255; return r;
    };
    float*  xr    = (float*)alloc(sizeof(float) * BDIM * CMIDP * HW);
    float*  offs  = (float*)alloc(sizeof(float) * BDIM * COFFP * HW);
    float*  fw    = (float*)alloc(sizeof(float) * BDIM * CMIDP * HW);
    float*  fh    = (float*)alloc(sizeof(float) * BDIM * CMIDP * HW);
    float*  favg  = (float*)alloc(sizeof(float) * BDIM * CMID_C);
    float*  attn  = (float*)alloc(sizeof(float) * BDIM * 2);
    __bf16* patches = (__bf16*)alloc(sizeof(__bf16) * BDIM * (size_t)KDEFP * HW);
    __bf16* wredP = (__bf16*)alloc(sizeof(__bf16) * CMIDP * CIN_C);
    __bf16* woffP = (__bf16*)alloc(sizeof(__bf16) * COFFP * CMIDP);
    __bf16* wdwP  = (__bf16*)alloc(sizeof(__bf16) * CMIDP * KDEFP);
    __bf16* wdhP  = (__bf16*)alloc(sizeof(__bf16) * CMIDP * KDEFP);
    __bf16* wexP  = (__bf16*)alloc(sizeof(__bf16) * CMIDP * CMIDP);
    float*  bredP = (float*)alloc(sizeof(float) * CMIDP);
    float*  boffP = (float*)alloc(sizeof(float) * COFFP);
    float*  bdwP  = (float*)alloc(sizeof(float) * CMIDP);
    float*  bdhP  = (float*)alloc(sizeof(float) * CMIDP);
    float*  bexP  = (float*)alloc(sizeof(float) * CMIDP);

    // ---- pack weights/biases into zero-padded bf16 ----
    pack_weight<<<ceil_div((long)CMIDP * CIN_C, 256), 256, 0, stream>>>(
        w_reduce, b_reduce, wredP, bredP, CMID_C, CIN_C, CMIDP, CIN_C);
    pack_weight<<<ceil_div((long)COFFP * CMIDP, 256), 256, 0, stream>>>(
        w_offset, b_offset, woffP, boffP, 2 * CMID_C, CMID_C, COFFP, CMIDP);
    pack_weight<<<ceil_div((long)CMIDP * KDEFP, 256), 256, 0, stream>>>(
        w_dcnw, b_dcnw, wdwP, bdwP, CMID_C, KDEF, CMIDP, KDEFP);
    pack_weight<<<ceil_div((long)CMIDP * KDEFP, 256), 256, 0, stream>>>(
        w_dcnh, b_dcnh, wdhP, bdhP, CMID_C, KDEF, CMIDP, KDEFP);
    pack_weight<<<ceil_div((long)CMIDP * CMIDP, 256), 256, 0, stream>>>(
        w_expand, b_expand, wexP, bexP, CIN_C, CMID_C, CMIDP, CMIDP);

    // 1) xr = conv1x1(x, w_reduce) + b_reduce      (MP=192, KP=192, B=x f32)
    {
        constexpr int waves = BDIM * (CMIDP / 16) * (HW / 64);
        wmma_gemm<CMIDP, CIN_C, 0><<<ceil_div(waves, 8), 256, 0, stream>>>(
            wredP, x, nullptr, nullptr, nullptr, bredP, xr, (long)CIN_C * HW);
    }
    // 2) offs = conv1x1(xr, w_offset) + b_offset   (MP=368, KP=192, B=xr f32)
    {
        constexpr int waves = BDIM * (COFFP / 16) * (HW / 64);
        wmma_gemm<COFFP, CMIDP, 0><<<ceil_div(waves, 8), 256, 0, stream>>>(
            woffP, xr, nullptr, nullptr, nullptr, boffP, offs, (long)CMIDP * HW);
    }
    // 3) f_w: im2col (off_w) then async-LDS deform GEMM
    {
        const long nP = (long)BDIM * KDEFP * HW;
        build_patches<<<ceil_div(nP, 256), 256, 0, stream>>>(xr, offs, 0, patches);
        constexpr int blocks = BDIM * (CMIDP / 64) * (HW / 128);   // 432
        wmma_gemm_deform<<<blocks, 256, 0, stream>>>(wdwP, patches, bdwP, fw);
    }
    // 4) f_h: im2col (off_h) then async-LDS deform GEMM (reuse patch buffer)
    {
        const long nP = (long)BDIM * KDEFP * HW;
        build_patches<<<ceil_div(nP, 256), 256, 0, stream>>>(xr, offs, CMID_C, patches);
        constexpr int blocks = BDIM * (CMIDP / 64) * (HW / 128);
        wmma_gemm_deform<<<blocks, 256, 0, stream>>>(wdhP, patches, bdhP, fh);
    }
    // 5) f_avg, attention
    mean_fwfh<<<BDIM * CMID_C, 256, 0, stream>>>(fw, fh, favg);
    attn_kernel<<<1, 64, 0, stream>>>(favg, w_attn, b_attn, attn);

    // 6) out = conv1x1(attn0*f_h + attn1*f_w, w_expand) + b_expand (fused)
    {
        constexpr int waves = BDIM * (CMIDP / 16) * (HW / 64);
        wmma_gemm<CMIDP, CMIDP, 2><<<ceil_div(waves, 8), 256, 0, stream>>>(
            wexP, nullptr, fh, fw, attn, bexP, out, (long)CMIDP * HW);
    }
    (void)in_sizes; (void)n_in; (void)out_size; (void)ws_size;
}
